// NeuTraLAD_31662498906974
// MI455X (gfx1250) — matrix operand order repaired and checked by reading, not compiled, not run on hardware
//
#include <hip/hip_runtime.h>
#include <hip/hip_bf16.h>
#include <math.h>

// Problem constants (match the reference)
static constexpr int Bn = 65536;   // batch
static constexpr int Dn = 256;     // input dim
static constexpr int Hn = 256;     // hidden dim
static constexpr int Kn = 11;      // number of transforms
static constexpr int NT = 256;     // threads per block (8 wave32)
static constexpr int ROWS = 128;   // batch rows per workgroup tile
static constexpr int LDP = 264;    // padded LDS row stride in bf16 elems (528B -> bank-conflict free)

typedef __attribute__((ext_vector_type(16))) __bf16 v16bf;
typedef __attribute__((ext_vector_type(8)))  float  v8f;

// ---------------------------------------------------------------------------
// WMMA fragment loads from LDS (layouts per CDNA5 ISA 7.12.2, bf16 16x16x32).
// A (16x32, MxK): lanes 0-15 hold K=[0..8)+[16..24), lanes 16-31 K=[8..16)+[24..32).
// B (32x16, KxN): stored transposed [N][K]; lanes 0-15 hold K=[0..8)+[16..24), lanes 16-31 K=[8..16)+[24..32).
// ---------------------------------------------------------------------------
__device__ __forceinline__ v16bf afrag(const __bf16* act, int mtile, int kk, int lane) {
  const int row  = mtile * 16 + (lane & 15);
  const int half = lane >> 4;
  const __bf16* p = act + row * LDP + kk * 32 + half * 8;
  union { uint4 u[2]; v16bf v; } t;
  t.u[0] = *(const uint4*)(p);        // K .. K+8
  t.u[1] = *(const uint4*)(p + 16);   // K+16 .. K+24
  return t.v;
}

__device__ __forceinline__ v16bf bfrag(const __bf16* wt, int ntile, int kk, int lane) {
  const int row = ntile * 16 + (lane & 15);
  const __bf16* p = wt + row * LDP + kk * 32 + (lane >> 4) * 8;
  union { uint4 u[2]; v16bf v; } t;
  t.u[0] = *(const uint4*)(p);
  t.u[1] = *(const uint4*)(p + 16);
  return t.v;
}

// Cooperative global(bf16 row-major, stride 256) -> LDS (stride LDP) copy.
__device__ __forceinline__ void copy_to_lds(const __bf16* __restrict__ g,
                                            __bf16* lds, int rows) {
  const int total = rows * 32;                    // uint4 chunks (8 bf16 each)
  for (int i = threadIdx.x; i < total; i += NT) {
    const int r = i >> 5, c8 = i & 31;
    *(uint4*)(lds + r * LDP + c8 * 8) = *(const uint4*)(g + r * 256 + c8 * 8);
  }
}

__device__ __forceinline__ float gelu_exact(float v) {
  return 0.5f * v * (1.0f + erff(v * 0.70710678118654752f));
}

// One GEMM stage: dst[ROWS x 256] = act(src[ROWS x 256] @ W[256 x 256] + bias)
// src: LDS bf16 (stride LDP); wt: LDS bf16 transposed [N][K] (stride LDP);
// dst: bf16, stride dstStride (LDS or global). Each wave owns M-stripe = wid.
// N is processed in two halves of 8 tiles to keep live accumulators at 64 VGPRs
// (a 16-tile accumulator block was observed to spill to scratch).
template <bool GELU>
__device__ __forceinline__ void gemm_stage(const __bf16* src, const __bf16* wt,
                                           const float* __restrict__ bias,
                                           __bf16* dst, int dstStride,
                                           int wid, int lane) {
  const int colBase = lane & 15;
  const int rofs    = (lane >> 4) * 8;
#pragma unroll 1
  for (int nh = 0; nh < 2; ++nh) {
    v8f acc[8] = {};
#pragma unroll 1
    for (int kk = 0; kk < 8; ++kk) {           // K = 256 = 8 x 32
      const v16bf a = afrag(src, wid, kk, lane);
#pragma unroll
      for (int n = 0; n < 8; ++n) {            // 8 N-tiles per half
        const v16bf b = bfrag(wt, nh * 8 + n, kk, lane);
        acc[n] = __builtin_amdgcn_wmma_f32_16x16x32_bf16(
            false, a, false, b, (short)0, acc[n], false, false);
      }
    }
    // Epilogue: C layout (ISA 7.12.2): VGPR j, lanes 0-15 -> M=j, lanes 16-31 -> M=j+8.
#pragma unroll
    for (int n = 0; n < 8; ++n) {
      const int col = (nh * 8 + n) * 16 + colBase;
      const float bv = bias[col];
#pragma unroll
      for (int j = 0; j < 8; ++j) {
        float v = acc[n][j] + bv;
        if (GELU) v = gelu_exact(v);
        const int row = wid * 16 + rofs + j;
        dst[(size_t)row * dstStride + col] = (__bf16)v;
      }
    }
  }
}

// ---------------------------------------------------------------------------
// Prep kernels
// ---------------------------------------------------------------------------
__global__ void f32_to_bf16_kernel(const float* __restrict__ s,
                                   __bf16* __restrict__ d, size_t n) {
  size_t i = ((size_t)blockIdx.x * NT + threadIdx.x) * 4;
  if (i + 3 < n) {
    const float4 v = *(const float4*)(s + i);
    d[i + 0] = (__bf16)v.x; d[i + 1] = (__bf16)v.y;
    d[i + 2] = (__bf16)v.z; d[i + 3] = (__bf16)v.w;
  }
}

// dst[k][c][r] = (bf16) src[k][r][c]  (weight transpose into WMMA-B layout)
__global__ void transpose_bf16_kernel(const float* __restrict__ src,
                                      __bf16* __restrict__ dst,
                                      int R, int C, int count) {
  const size_t per = (size_t)R * C;
  const size_t idx = (size_t)blockIdx.x * NT + threadIdx.x;
  if (idx < per * (size_t)count) {
    const size_t k = idx / per, rc = idx % per;
    const int r = (int)(rc / C), c = (int)(rc % C);
    dst[k * per + (size_t)c * R + r] = (__bf16)src[idx];
  }
}

// ---------------------------------------------------------------------------
// Fused 4-GEMM chain: zt_k = enc(transform_k(x)) ; or z = enc(x) for k == Kn.
// Dynamic LDS: actA | actB (ROWS x LDP) | W (256 x LDP)  = 270336 bytes.
// ---------------------------------------------------------------------------
__global__ void __launch_bounds__(NT, 1)
chain_kernel(const __bf16* __restrict__ xbf,
             const __bf16* __restrict__ w1t,
             const __bf16* __restrict__ w2t,
             const __bf16* __restrict__ ew1t,
             const __bf16* __restrict__ ew2t,
             const float* __restrict__ Tb1,
             const float* __restrict__ Tb2,
             const float* __restrict__ Eb1,
             const float* __restrict__ Eb2,
             __bf16* __restrict__ zt,
             __bf16* __restrict__ z) {
  extern __shared__ __bf16 smem[];
  __bf16* A  = smem;                 // ROWS x LDP
  __bf16* Bt = A + ROWS * LDP;       // ROWS x LDP
  __bf16* W  = Bt + ROWS * LDP;      // 256  x LDP

  const int blockRow = blockIdx.x;
  const int k        = blockIdx.y;
  const int wid  = threadIdx.x >> 5;
  const int lane = threadIdx.x & 31;

  copy_to_lds(xbf + (size_t)blockRow * ROWS * 256, A, ROWS);

  if (k < Kn) {
    const __bf16* tw1 = w1t + (size_t)k * Dn * Hn;
    const __bf16* tw2 = w2t + (size_t)k * Hn * Dn;
    copy_to_lds(tw1, W, 256);
    __builtin_prefetch(tw2 + (size_t)threadIdx.x * 256, 0, 2);   // next weights -> L2
    __syncthreads();
    gemm_stage<true >(A, W, Tb1 + k * Hn, Bt, LDP, wid, lane);   // gelu(x@Tw1+b1)
    __syncthreads();

    copy_to_lds(tw2, W, 256);
    __builtin_prefetch(ew1t + (size_t)threadIdx.x * 256, 0, 2);
    __syncthreads();
    gemm_stage<false>(Bt, W, Tb2 + k * Dn, A, LDP, wid, lane);   // tx = h@Tw2+b2 (overwrite x tile)
    __syncthreads();

    copy_to_lds(ew1t, W, 256);
    __builtin_prefetch(ew2t + (size_t)threadIdx.x * 256, 0, 2);
    __syncthreads();
    gemm_stage<true >(A, W, Eb1, Bt, LDP, wid, lane);            // gelu(tx@Ew1+b)
    __syncthreads();

    copy_to_lds(ew2t, W, 256);
    __syncthreads();
    __bf16* dst = zt + ((size_t)k * Bn + (size_t)blockRow * ROWS) * 256;
    gemm_stage<false>(Bt, W, Eb2, dst, 256, wid, lane);          // zt_k -> global
  } else {
    copy_to_lds(ew1t, W, 256);
    __builtin_prefetch(ew2t + (size_t)threadIdx.x * 256, 0, 2);
    __syncthreads();
    gemm_stage<true >(A, W, Eb1, Bt, LDP, wid, lane);
    __syncthreads();
    copy_to_lds(ew2t, W, 256);
    __syncthreads();
    gemm_stage<false>(Bt, W, Eb2, z + (size_t)blockRow * ROWS * 256, 256, wid, lane);
  }
}

// ---------------------------------------------------------------------------
// Scoring: one wave32 per batch row. 12 vectors (z, zt_1..11) x 256 bf16.
// Processed in two half-passes (4 f32 per lane live) accumulating a
// triangular Gram g[78]; one shuffle-reduction at the end.
// ---------------------------------------------------------------------------
__device__ __forceinline__ void load4f(const __bf16* p, float* o) {
  union { uint2 u; __bf16 h[4]; } t;
  t.u = *(const uint2*)p;
#pragma unroll
  for (int i = 0; i < 4; ++i) o[i] = (float)t.h[i];
}

// triangular index for i <= j over 12 vectors (78 entries)
__host__ __device__ constexpr int tix(int i, int j) {
  return 12 * i - (i * (i - 1)) / 2 + (j - i);
}

__global__ void __launch_bounds__(NT, 1)
score_kernel(const __bf16* __restrict__ z,
             const __bf16* __restrict__ zt,
             float* __restrict__ out) {
  const int wid  = threadIdx.x >> 5;
  const int lane = threadIdx.x & 31;
  const size_t b = (size_t)blockIdx.x * 8 + wid;

  float g[78];
#pragma unroll
  for (int i = 0; i < 78; ++i) g[i] = 0.f;

#pragma unroll 1
  for (int half = 0; half < 2; ++half) {
    const int off = half * 128 + lane * 4;
    float v[12][4];
    load4f(z + b * 256 + off, v[0]);
#pragma unroll
    for (int k = 0; k < Kn; ++k)
      load4f(zt + ((size_t)k * Bn + b) * 256 + off, v[k + 1]);

#pragma unroll
    for (int i = 0; i < 12; ++i)
#pragma unroll
      for (int j = i; j < 12; ++j) {
        float s = g[tix(i, j)];
#pragma unroll
        for (int e = 0; e < 4; ++e) s += v[i][e] * v[j][e];
        g[tix(i, j)] = s;
      }
  }

  // wave32 reduction of all 78 partial dots
#pragma unroll
  for (int i = 0; i < 78; ++i) {
    float s = g[i];
#pragma unroll
    for (int o = 16; o > 0; o >>= 1) s += __shfl_xor(s, o, 32);
    g[i] = s;
  }

  const float EPS = 1e-8f;
  float nrm[12];
#pragma unroll
  for (int i = 0; i < 12; ++i) nrm[i] = sqrtf(g[tix(i, i)]);

  float total = 0.f;
#pragma unroll
  for (int k = 1; k <= Kn; ++k) {
    const float sz = expf(g[tix(0, k)] / fmaxf(nrm[0] * nrm[k], EPS));  // TEMP = 1
    float neg = 0.f;
#pragma unroll
    for (int l = 1; l <= Kn; ++l)
      if (l != k) {
        const int i = l < k ? l : k, j = l < k ? k : l;
        neg += expf(g[tix(i, j)] / fmaxf(nrm[l] * nrm[k], EPS));
      }
    total += logf(sz / (sz + neg));
  }
  if (lane == 0) out[b] = -total;
}

// ---------------------------------------------------------------------------
// Host launch
// ---------------------------------------------------------------------------
extern "C" void kernel_launch(void* const* d_in, const int* in_sizes, int n_in,
                              void* d_out, int out_size, void* d_ws, size_t ws_size,
                              hipStream_t stream) {
  const float* x   = (const float*)d_in[0];
  const float* Tw1 = (const float*)d_in[1];
  const float* Tb1 = (const float*)d_in[2];
  const float* Tw2 = (const float*)d_in[3];
  const float* Tb2 = (const float*)d_in[4];
  const float* Ew1 = (const float*)d_in[5];
  const float* Eb1 = (const float*)d_in[6];
  const float* Ew2 = (const float*)d_in[7];
  const float* Eb2 = (const float*)d_in[8];
  float* out = (float*)d_out;

  // Workspace partition (bf16)
  __bf16* p    = (__bf16*)d_ws;
  __bf16* xbf  = p; p += (size_t)Bn * Dn;        // 16M
  __bf16* zbuf = p; p += (size_t)Bn * Hn;        // 16M
  __bf16* ztb  = p; p += (size_t)Kn * Bn * Hn;   // 184M
  __bf16* w1t  = p; p += (size_t)Kn * Dn * Hn;
  __bf16* w2t  = p; p += (size_t)Kn * Hn * Dn;
  __bf16* ew1t = p; p += (size_t)Dn * Hn;
  __bf16* ew2t = p; p += (size_t)Hn * Hn;

  // --- prep: convert x; transpose+convert weights into [N][K] bf16 layout ---
  {
    const size_t n = (size_t)Bn * Dn;
    f32_to_bf16_kernel<<<(unsigned)(n / (NT * 4)), NT, 0, stream>>>(x, xbf, n);
  }
  {
    const int per = Dn * Hn;
    unsigned gk = (unsigned)(((size_t)Kn * per + NT - 1) / NT);
    unsigned g1 = (unsigned)((per + NT - 1) / NT);
    transpose_bf16_kernel<<<gk, NT, 0, stream>>>(Tw1, w1t, Dn, Hn, Kn);
    transpose_bf16_kernel<<<gk, NT, 0, stream>>>(Tw2, w2t, Hn, Dn, Kn);
    transpose_bf16_kernel<<<g1, NT, 0, stream>>>(Ew1, ew1t, Dn, Hn, 1);
    transpose_bf16_kernel<<<g1, NT, 0, stream>>>(Ew2, ew2t, Hn, Hn, 1);
  }

  // --- fused transform+encoder chains (WMMA bf16) ---
  {
    const size_t ldsBytes = (size_t)(2 * ROWS + 256) * LDP * sizeof(__bf16); // 270336
    dim3 grid(Bn / ROWS, Kn + 1);
    chain_kernel<<<grid, NT, ldsBytes, stream>>>(xbf, w1t, w2t, ew1t, ew2t,
                                                 Tb1, Tb2, Eb1, Eb2, ztb, zbuf);
  }

  // --- scoring ---
  score_kernel<<<Bn / 8, NT, 0, stream>>>(zbuf, ztb, out);
}
